// LinearAttentionLayer_55551107006708
// MI455X (gfx1250) — matrix operand order repaired and checked by reading, not compiled, not run on hardware
//
#include <hip/hip_runtime.h>
#include <hip/hip_bf16.h>
#include <math.h>

// ---------------------------------------------------------------------------
// CDNA5 (gfx1250) wave32 WMMA implementation of a causal linear-attention
// layer:  qkv GEMM -> chunked linear attention (scan as matmuls) -> out GEMM.
// Matmuls use v_wmma_f32_16x16x32_f16 (f32 accumulate); f16 workspace tiles
// are staged into LDS with GLOBAL_LOAD_ASYNC_TO_LDS_B128 (ASYNCcnt-tracked).
// All staging loops have fixed trip counts (no per-iteration exec guards).
// ---------------------------------------------------------------------------

typedef __attribute__((ext_vector_type(16))) _Float16 v16h;
typedef __attribute__((ext_vector_type(8)))  _Float16 h8;
typedef __attribute__((ext_vector_type(4)))  _Float16 h4;
typedef __attribute__((ext_vector_type(8)))  float    v8f;
typedef int i128 __attribute__((vector_size(16)));   // 16-byte DMA granule

#define HEADS 16
#define DH    64
#define DIM   1024
#define NTOK  2048
#define NQKV  3072

__device__ __forceinline__ v8f wmma_f16(v16h a, v16h b, v8f c) {
    return __builtin_amdgcn_wmma_f32_16x16x32_f16(
        /*neg_a=*/false, a, /*neg_b=*/false, b,
        /*c_mod=*/(short)0, c, /*reuse_a=*/false, /*reuse_b=*/false);
}

// --- async global->LDS copy (16 bytes per lane), ASYNCcnt-tracked -----------
__device__ __forceinline__ void async_cp_b128(const void* gsrc, void* ldst) {
#if __has_builtin(__builtin_amdgcn_global_load_async_to_lds_b128)
    __builtin_amdgcn_global_load_async_to_lds_b128(
        (__attribute__((address_space(1))) i128*)gsrc,
        (__attribute__((address_space(3))) i128*)ldst, 0, 0);
#else
    *(h8*)ldst = *(const h8*)gsrc;
#endif
}
__device__ __forceinline__ void async_wait() {
#if __has_builtin(__builtin_amdgcn_global_load_async_to_lds_b128)
  #if __has_builtin(__builtin_amdgcn_s_wait_asynccnt)
    __builtin_amdgcn_s_wait_asynccnt(0);
  #else
    asm volatile("s_wait_asynccnt 0x0" ::: "memory");
  #endif
#endif
}

// A fragment: logical A[16 x 32] f16 row-major in LDS (`stride` elems/row).
// ISA layout: lanes 0-15 -> K {0..7,16..23}, lanes 16-31 -> K {8..15,24..31}.
__device__ __forceinline__ v16h load_frag_A(const _Float16* p0, int stride,
                                            int row_base, int lane) {
    const int half = lane >> 4;
    const _Float16* p = p0 + (row_base + (lane & 15)) * stride;
    v16h a;
#pragma unroll
    for (int j = 0; j < 8; ++j) {
        int k = ((j < 4) ? (2 * j) : (16 + 2 * (j - 4))) + half * 8;
        a[2 * j]     = p[k];
        a[2 * j + 1] = p[k + 1];
    }
    return a;
}

// A fragment from a transposed LDS image: logical A[row][k] = p0[k*stride+row].
__device__ __forceinline__ v16h load_frag_A_tr(const _Float16* p0, int stride,
                                               int row_base, int lane) {
    const int half = lane >> 4;
    const int row  = row_base + (lane & 15);
    v16h a;
#pragma unroll
    for (int j = 0; j < 8; ++j) {
        int k = ((j < 4) ? (2 * j) : (16 + 2 * (j - 4))) + half * 8;
        a[2 * j]     = p0[k * stride + row];
        a[2 * j + 1] = p0[(k + 1) * stride + row];
    }
    return a;
}

// B fragment: logical B[32 x 16] staged TRANSPOSED in LDS as Bt[col][k].
__device__ __forceinline__ v16h load_frag_B(const _Float16* p0, int stride,
                                            int col_base, int lane) {
    const int half = lane >> 4;
    const _Float16* p = p0 + (col_base + (lane & 15)) * stride + half * 8;
    v16h b;
#pragma unroll
    for (int j = 0; j < 16; ++j) b[j] = p[(j < 8) ? j : (j + 8)];
    return b;
}

// ---------------------------------------------------------------------------
// Kernel 1: QKV = X @ W_qkv, fused feature map (elu+1) on q/k, mask on k/v.
// Writes f16 workspace: q_ws,k_ws as [H][n][DH]; v transposed [H][DH][n].
// Block = 128 threads (4 waves); tile = 64M x 128N; K-step = 32.
// ---------------------------------------------------------------------------
__global__ __launch_bounds__(128)
void la_qkv_gemm(const float* __restrict__ X, const float* __restrict__ W,
                 const unsigned char* __restrict__ mask,
                 _Float16* __restrict__ q_ws, _Float16* __restrict__ k_ws,
                 _Float16* __restrict__ vT_ws) {
    __shared__ _Float16 As[64 * 32];    // [row][k]
    __shared__ _Float16 Bs[128 * 32];   // transposed: [col][k]
    const int n0   = blockIdx.x * 128;
    const int m0   = blockIdx.y * 64;
    const int tid  = threadIdx.x;
    const int wave = tid >> 5, lane = tid & 31;
    const int half = lane >> 4, l16 = lane & 15;

    // fixed per-thread staging coordinates
    const int ar = tid >> 3, akq = (tid & 7) * 4;        // A: 64x32, 16 rows/pass
    const int bk = tid >> 2, bcq = (tid & 3) * 4;        // B: 32x128, 8 passes

    v8f acc[8] = {};

    for (int k0 = 0; k0 < DIM; k0 += 32) {
        // phase 1: issue all staging loads (no waits in between)
        float4 xa[4], wb[8];
#pragma unroll
        for (int it = 0; it < 4; ++it)   // A rows ar, ar+16, ar+32, ar+48
            xa[it] = *(const float4*)&X[(size_t)(m0 + ar + it * 16) * DIM + k0 + akq];
#pragma unroll
        for (int it = 0; it < 8; ++it)   // B cols bcq + it*16
            wb[it] = *(const float4*)&W[(size_t)(k0 + bk) * NQKV + n0 + bcq + it * 16];
        if (k0 + 32 < DIM) {             // speculative prefetch of next K tile
            __builtin_prefetch(&X[(size_t)(m0 + (tid >> 1)) * DIM + k0 + 32], 0, 0);
            __builtin_prefetch(&W[(size_t)(k0 + 32 + (tid >> 2)) * NQKV + n0], 0, 0);
        }
        // phase 2: convert + LDS store
#pragma unroll
        for (int it = 0; it < 4; ++it) {
            h4 hv = {(_Float16)xa[it].x, (_Float16)xa[it].y,
                     (_Float16)xa[it].z, (_Float16)xa[it].w};
            *(h4*)&As[(ar + it * 16) * 32 + akq] = hv;
        }
#pragma unroll
        for (int it = 0; it < 8; ++it) {
            const int c = bcq + it * 16;
            Bs[(c + 0) * 32 + bk] = (_Float16)wb[it].x;
            Bs[(c + 1) * 32 + bk] = (_Float16)wb[it].y;
            Bs[(c + 2) * 32 + bk] = (_Float16)wb[it].z;
            Bs[(c + 3) * 32 + bk] = (_Float16)wb[it].w;
        }
        __syncthreads();

        v16h a = load_frag_A(As, 32, wave * 16, lane);
#pragma unroll
        for (int ct = 0; ct < 8; ++ct) {
            v16h b = load_frag_B(Bs, 32, ct * 16, lane);
            acc[ct] = wmma_f16(a, b, acc[ct]);
        }
        __syncthreads();
    }

    // epilogue: feature map + mask + scatter to head-major f16 ws
#pragma unroll
    for (int ct = 0; ct < 8; ++ct) {
        const int gf   = n0 + ct * 16 + l16;     // column in [0,3072)
        const int part = gf >> 10;               // 0=q, 1=k, 2=v
        const int hd   = gf & 1023;
        const int h    = hd >> 6;
        const int d    = hd & 63;
#pragma unroll
        for (int v = 0; v < 8; ++v) {
            const int t = m0 + wave * 16 + v + 8 * half;  // token index
            float val = acc[ct][v];
            if (part <= 1) val = (val > 0.f) ? (val + 1.f) : __expf(val);
            if (part >= 1) val *= (mask[t] ? 1.f : 0.f);
            if (part == 0)
                q_ws[(size_t)(h * NTOK + t) * DH + d] = (_Float16)val;
            else if (part == 1)
                k_ws[(size_t)(h * NTOK + t) * DH + d] = (_Float16)val;
            else
                vT_ws[(size_t)(h * DH + d) * NTOK + t] = (_Float16)val;
        }
    }
}

// ---------------------------------------------------------------------------
// Kernel 2: chunked causal linear attention.  One workgroup per head, serial
// over 32 chunks of 64 tokens; Q/K/V tiles staged via async global->LDS DMA.
//   A1  = causal_mask(Q Kt)                 (WMMA)
//   out = (A1 @ V + Q @ S_prev) / den       (WMMA)
//   S  += Kt @ V ; z += colsum(K)           (WMMA)
//   den = rowsum(A1) + Q . z_prev + eps
// ---------------------------------------------------------------------------
__global__ __launch_bounds__(128)
void la_linattn(const _Float16* __restrict__ q_ws,
                const _Float16* __restrict__ k_ws,
                const _Float16* __restrict__ vT_ws,
                _Float16* __restrict__ att) {
    const int h = blockIdx.x;
    __shared__ _Float16 Qc[64 * 64];    // [tok][d]
    __shared__ _Float16 Kc[64 * 64];    // [tok][d]  == Bt(col=tok,k=d)
    __shared__ _Float16 Vt[64 * 64];    // [d][tok]  == Bt(col=d,k=tok)
    __shared__ _Float16 A1s[64 * 64];   // masked scores [tok][s]
    __shared__ _Float16 StT[64 * 64];   // S^T staged f16: [e][d]
    __shared__ float    Sst[64 * 64];   // state S fp32: [d][e]
    __shared__ float    zst[64];
    __shared__ float    den[64];

    const int tid  = threadIdx.x;
    const int wave = tid >> 5, lane = tid & 31;
    const int half = lane >> 4, l16 = lane & 15;
    const int rb   = wave * 16;
    const int vd   = tid >> 3, vcq = (tid & 7) * 8;   // V staging coords

#pragma unroll
    for (int it = 0; it < 32; ++it) Sst[it * 128 + tid] = 0.f;
    if (tid < 64) zst[tid] = 0.f;
    __syncthreads();

    const _Float16* qh = q_ws + (size_t)h * NTOK * DH;
    const _Float16* kh = k_ws + (size_t)h * NTOK * DH;
    const _Float16* vh = vT_ws + (size_t)h * DH * NTOK;

    for (int base = 0; base < NTOK; base += 64) {
        // ---- async DMA of Q/K (contiguous 8KB) and V (row-sliced) ----
        const _Float16* qs = qh + (size_t)base * DH;
        const _Float16* ks = kh + (size_t)base * DH;
#pragma unroll
        for (int it = 0; it < 4; ++it) {
            const int e = (it * 128 + tid) * 8;
            async_cp_b128(qs + e, Qc + e);
            async_cp_b128(ks + e, Kc + e);
        }
#pragma unroll
        for (int it = 0; it < 4; ++it) {
            const int d = vd + it * 16;
            async_cp_b128(&vh[(size_t)d * NTOK + base + vcq], &Vt[d * 64 + vcq]);
        }
        // transposed f16 image of current fp32 state (i = row d, j = col e)
#pragma unroll
        for (int it = 0; it < 32; ++it) {
            const int idx = it * 128 + tid;
            const int i = idx >> 6, j = idx & 63;
            StT[j * 64 + i] = (_Float16)Sst[i * 64 + j];
        }
        async_wait();
        __syncthreads();

        // ---- GEMM1: A1 = mask(Q @ K^T) ----
        {
            v16h a0 = load_frag_A(Qc + 0, 64, rb, lane);
            v16h a1 = load_frag_A(Qc + 32, 64, rb, lane);
#pragma unroll
            for (int ct = 0; ct < 4; ++ct) {
                v8f acc = {};
                acc = wmma_f16(a0, load_frag_B(Kc + 0, 64, ct * 16, lane), acc);
                acc = wmma_f16(a1, load_frag_B(Kc + 32, 64, ct * 16, lane), acc);
                const int col = ct * 16 + l16;
#pragma unroll
                for (int v = 0; v < 8; ++v) {
                    const int row = rb + v + 8 * half;
                    float s = (col <= row) ? acc[v] : 0.f;   // causal mask
                    A1s[row * 64 + col] = (_Float16)s;
                }
            }
        }
        __syncthreads();

        // ---- denominator: rowsum(A1) + Q . z_prev + eps ----
        if (tid < 64) {
            float s = 1e-6f;
#pragma unroll 8
            for (int j = 0; j < 64; ++j) s += (float)A1s[tid * 64 + j];
#pragma unroll 8
            for (int d = 0; d < 64; ++d) s += (float)Qc[tid * 64 + d] * zst[d];
            den[tid] = s;
        }
        __syncthreads();

        // ---- GEMM2: out = (A1 @ V + Q @ S_prev) / den ----
        {
            v16h aq0 = load_frag_A(Qc + 0, 64, rb, lane);
            v16h aq1 = load_frag_A(Qc + 32, 64, rb, lane);
            v16h as0 = load_frag_A(A1s + 0, 64, rb, lane);
            v16h as1 = load_frag_A(A1s + 32, 64, rb, lane);
#pragma unroll
            for (int ct = 0; ct < 4; ++ct) {
                v8f acc = {};
                acc = wmma_f16(as0, load_frag_B(Vt + 0, 64, ct * 16, lane), acc);
                acc = wmma_f16(as1, load_frag_B(Vt + 32, 64, ct * 16, lane), acc);
                acc = wmma_f16(aq0, load_frag_B(StT + 0, 64, ct * 16, lane), acc);
                acc = wmma_f16(aq1, load_frag_B(StT + 32, 64, ct * 16, lane), acc);
                const int dcol = ct * 16 + l16;
#pragma unroll
                for (int v = 0; v < 8; ++v) {
                    const int row = rb + v + 8 * half;
                    att[(size_t)(base + row) * DIM + h * DH + dcol] =
                        (_Float16)(acc[v] / den[row]);
                }
            }
            // ---- state update: S += K^T @ V ----
            v16h ak0 = load_frag_A_tr(Kc, 64, rb, lane);            // A[d][s]
            v16h ak1 = load_frag_A_tr(Kc + 32 * 64, 64, rb, lane);
#pragma unroll
            for (int ct = 0; ct < 4; ++ct) {
                v8f acc = {};
                acc = wmma_f16(ak0, load_frag_B(Vt + 0, 64, ct * 16, lane), acc);
                acc = wmma_f16(ak1, load_frag_B(Vt + 32, 64, ct * 16, lane), acc);
                const int ecol = ct * 16 + l16;
#pragma unroll
                for (int v = 0; v < 8; ++v) {
                    const int drow = rb + v + 8 * half;
                    Sst[drow * 64 + ecol] += acc[v];
                }
            }
        }
        // ---- z += colsum(K) (old zst already consumed by den) ----
        if (tid < 64) {
            float s = zst[tid];
#pragma unroll 8
            for (int r = 0; r < 64; ++r) s += (float)Kc[r * 64 + tid];
            zst[tid] = s;
        }
        __syncthreads();
    }
}

// ---------------------------------------------------------------------------
// Kernel 3: out = att @ W_out + b_out   (2048x1024 @ 1024x1024)
// att is f16 -> A tile staged with async DMA; W converted f32->f16.
// Tile = 64M x 128N.
// ---------------------------------------------------------------------------
__global__ __launch_bounds__(128)
void la_out_gemm(const _Float16* __restrict__ att, const float* __restrict__ W,
                 const float* __restrict__ bias, float* __restrict__ out) {
    __shared__ _Float16 As[64 * 32];
    __shared__ _Float16 Bs[128 * 32];   // transposed [col][k]
    const int n0   = blockIdx.x * 128;
    const int m0   = blockIdx.y * 64;
    const int tid  = threadIdx.x;
    const int wave = tid >> 5, lane = tid & 31;
    const int half = lane >> 4, l16 = lane & 15;

    const int ar = tid >> 1, acq = (tid & 1) * 8;     // A: 64 rows x 4 chunks
    const int bk = tid >> 2, bcq = (tid & 3) * 4;     // B: 32x128

    v8f acc[8] = {};

    for (int k0 = 0; k0 < DIM; k0 += 32) {
        // A tile 64x32 f16: pure async DMA (2 b128 chunks per thread)
#pragma unroll
        for (int it = 0; it < 2; ++it) {
            const int cq = acq + it * 16;
            async_cp_b128(&att[(size_t)(m0 + ar) * DIM + k0 + cq],
                          &As[ar * 32 + cq]);
        }
        // B tile 32x128 (transposed write): float4 along columns
        float4 wb[8];
#pragma unroll
        for (int it = 0; it < 8; ++it)
            wb[it] = *(const float4*)&W[(size_t)(k0 + bk) * DIM + n0 + bcq + it * 16];
        if (k0 + 32 < DIM)
            __builtin_prefetch(&W[(size_t)(k0 + 32 + (tid >> 2)) * DIM + n0], 0, 0);
#pragma unroll
        for (int it = 0; it < 8; ++it) {
            const int c = bcq + it * 16;
            Bs[(c + 0) * 32 + bk] = (_Float16)wb[it].x;
            Bs[(c + 1) * 32 + bk] = (_Float16)wb[it].y;
            Bs[(c + 2) * 32 + bk] = (_Float16)wb[it].z;
            Bs[(c + 3) * 32 + bk] = (_Float16)wb[it].w;
        }
        async_wait();
        __syncthreads();

        v16h a = load_frag_A(As, 32, wave * 16, lane);
#pragma unroll
        for (int ct = 0; ct < 8; ++ct) {
            v16h b = load_frag_B(Bs, 32, ct * 16, lane);
            acc[ct] = wmma_f16(a, b, acc[ct]);
        }
        __syncthreads();
    }

#pragma unroll
    for (int ct = 0; ct < 8; ++ct) {
        const int gf = n0 + ct * 16 + l16;
        const float bv = bias[gf];
#pragma unroll
        for (int v = 0; v < 8; ++v) {
            const int t = m0 + wave * 16 + v + 8 * half;
            out[(size_t)t * DIM + gf] = acc[ct][v] + bv;
        }
    }
}

// ---------------------------------------------------------------------------
extern "C" void kernel_launch(void* const* d_in, const int* in_sizes, int n_in,
                              void* d_out, int out_size, void* d_ws,
                              size_t ws_size, hipStream_t stream) {
    const float*         q      = (const float*)d_in[0];
    const unsigned char* mask   = (const unsigned char*)d_in[1];
    const float*         w_qkv  = (const float*)d_in[2];
    const float*         w_out  = (const float*)d_in[3];
    const float*         b_out  = (const float*)d_in[4];
    float*               out    = (float*)d_out;

    const size_t perbuf = (size_t)HEADS * NTOK * DH;  // 2M halfs
    _Float16* ws    = (_Float16*)d_ws;
    _Float16* q_ws  = ws;
    _Float16* k_ws  = ws + perbuf;
    _Float16* vT_ws = ws + 2 * perbuf;
    _Float16* att   = ws + 3 * perbuf;

    la_qkv_gemm<<<dim3(NQKV / 128, NTOK / 64), 128, 0, stream>>>(
        q, w_qkv, mask, q_ws, k_ws, vT_ws);
    la_linattn<<<dim3(HEADS), 128, 0, stream>>>(q_ws, k_ws, vT_ws, att);
    la_out_gemm<<<dim3(DIM / 128, NTOK / 64), 128, 0, stream>>>(
        att, w_out, b_out, out);
}
